// Spatial_Attention_31104153158199
// MI455X (gfx1250) — matrix operand order repaired and checked by reading, not compiled, not run on hardware
//
#include <hip/hip_runtime.h>

// ---------------------------------------------------------------------------
// Spatial attention forward for MI455X (gfx1250, wave32, WMMA bf16 16x16x32)
// B=8192, V=17, DIM=192, HEADS=3, DKH=64
// ---------------------------------------------------------------------------

typedef __attribute__((ext_vector_type(16))) __bf16 v16bf;
typedef __attribute__((ext_vector_type(8)))  __bf16 v8bf;
typedef __attribute__((ext_vector_type(4)))  __bf16 v4bf;
typedef __attribute__((ext_vector_type(8)))  float  v8f;

#define BATCH 8192
#define VN    17
#define DIMC  192
#define NH    3
#define DKH   64
#define MROWS (BATCH * VN)      // 139264, divisible by 64
#define QLD   392               // q,k LDS row (bf16): 384 data + pad; 784B (16B mult)

__device__ __forceinline__ v8f wmma_bf16(v16bf a, v16bf b, v8f c) {
  return __builtin_amdgcn_wmma_f32_16x16x32_bf16(false, a, false, b,
                                                 (short)0, c, false, false);
}

__device__ __forceinline__ v16bf combine16(v8bf lo, v8bf hi) {
  return __builtin_shufflevector(lo, hi, 0, 1, 2, 3, 4, 5, 6, 7,
                                 8, 9, 10, 11, 12, 13, 14, 15);
}

// A operand (16x32 bf16) from row-major LDS [m][k]; rows 16B-aligned.
__device__ __forceinline__ v16bf packA(const __bf16* base, int ld) {
  int lane = threadIdx.x & 31;
  int lh   = lane >> 4;
  const v8bf* row = (const v8bf*)(base + (lane & 15) * ld);
  return combine16(row[lh], row[2 + lh]);
}

// A operand with row padding (rows >= nrows read as 0)
__device__ __forceinline__ v16bf packA_pad(const __bf16* base, int ld, int row0, int nrows) {
  int lane = threadIdx.x & 31;
  int lh   = lane >> 4;
  int m    = row0 + (lane & 15);
  if (m < nrows) {
    const v8bf* row = (const v8bf*)(base + m * ld);
    return combine16(row[lh], row[2 + lh]);
  }
  v16bf z = {};
  return z;
}

// B operand (32x16 bf16) from B^T LDS layout Bt[n][k] (n-major, k-contiguous).
__device__ __forceinline__ v16bf packBt(const __bf16* base, int ld) {
  int lane = threadIdx.x & 31;
  int lh   = lane >> 4;
  const v8bf* row = (const v8bf*)(base + (lane & 15) * ld);
  return combine16(row[lh], row[2 + lh]);
}

// Same with n-row padding (rows >= nlimit read as 0)
__device__ __forceinline__ v16bf packBt_pad(const __bf16* base, int ld, int ntoff, int nlimit) {
  int lane = threadIdx.x & 31;
  int lh   = lane >> 4;
  int n    = ntoff + (lane & 15);
  if (n < nlimit) {
    const v8bf* row = (const v8bf*)(base + n * ld);
    return combine16(row[lh], row[2 + lh]);
  }
  v16bf z = {};
  return z;
}

__device__ __forceinline__ v4bf cvt4(float4 f) {
  v4bf o;
  o[0] = (__bf16)f.x; o[1] = (__bf16)f.y; o[2] = (__bf16)f.z; o[3] = (__bf16)f.w;
  return o;
}

// 64-elem dot: bf16 (LDS, 16B-aligned) x f32 (global)
__device__ __forceinline__ float dot64_bf(const __bf16* qr, const float* kr) {
  const v8bf* qv = (const v8bf*)qr;
  float s = 0.f;
#pragma unroll
  for (int c = 0; c < 8; ++c) {
    v8bf q8 = qv[c];
#pragma unroll
    for (int e = 0; e < 8; ++e) s += (float)q8[e] * kr[c * 8 + e];
  }
  return s;
}

// ---------------------------------------------------------------------------
// qkv GEMM: Out(M x N, bf16) = X(M x 192, f32) @ W(N x 192, f32)^T + bias
// A operands preloaded to registers; full 64x192 B panel staged per n-block.
// ---------------------------------------------------------------------------
__global__ __launch_bounds__(256) void gemm_qkv_kernel(
    const float* __restrict__ X, const float* __restrict__ W,
    const float* __restrict__ bias, __bf16* __restrict__ Out, int N) {
  __shared__ __bf16 As[64][200];   // [m][k]
  __shared__ __bf16 Bf[64][200];   // [n][k] full panel
  int tid = threadIdx.x;
  int m0  = blockIdx.x * 64;
#pragma unroll
  for (int i = 0; i < 12; ++i) {   // 64x192 f32 via float4
    int idx = tid + 256 * i;
    int m = idx / 48, c = idx - m * 48;
    float4 f = ((const float4*)(X + (size_t)(m0 + m) * 192))[c];
    *(v4bf*)&As[m][c * 4] = cvt4(f);
  }
  __syncthreads();
  int wave = tid >> 5, lane = tid & 31;
  int wm = wave >> 1, wn = wave & 1;
  int lh = (lane >> 4) & 1, nl = lane & 15;

  v16bf areg[6];
#pragma unroll
  for (int ks = 0; ks < 6; ++ks) areg[ks] = packA(&As[wm * 16][ks * 32], 200);

  int nblocks = N >> 6;
  for (int nb = 0; nb < nblocks; ++nb) {
    int n0 = nb * 64;
    __syncthreads();                  // previous Bf reads done
#pragma unroll
    for (int i = 0; i < 12; ++i) {   // 64x192 weight panel via float4
      int idx = tid + 256 * i;
      int n = idx / 48, kc = idx - n * 48;
      float4 f = ((const float4*)(W + (size_t)(n0 + n) * 192))[kc];
      *(v4bf*)&Bf[n][kc * 4] = cvt4(f);
    }
    __syncthreads();
    v8f acc0 = {}, acc1 = {};
#pragma unroll
    for (int ks = 0; ks < 6; ++ks) {
      acc0 = wmma_bf16(areg[ks], packBt(&Bf[wn * 32][ks * 32], 200), acc0);
      acc1 = wmma_bf16(areg[ks], packBt(&Bf[wn * 32 + 16][ks * 32], 200), acc1);
    }
    int gn = n0 + wn * 32 + nl;
    int gm = m0 + wm * 16 + lh * 8;
#pragma unroll
    for (int r = 0; r < 8; ++r) {
      size_t o = (size_t)(gm + r) * N;
      Out[o + gn]      = (__bf16)(acc0[r] + bias[gn]);
      Out[o + gn + 16] = (__bf16)(acc1[r] + bias[gn + 16]);
    }
  }
}

// ---------------------------------------------------------------------------
// Attention core: one workgroup (96 thr = 3 waves) per batch; wave h = head h.
// ---------------------------------------------------------------------------
__global__ __launch_bounds__(96) void attn_kernel(
    const __bf16* __restrict__ qkv, const float* __restrict__ mask,
    const float* __restrict__ PA, const float* __restrict__ gmask,
    const float* __restrict__ drop, const float* __restrict__ key_rel,
    const float* __restrict__ key_rel_diag, __bf16* __restrict__ outA) {
  __shared__ __bf16 Q[VN * QLD];        // q,k rows (cols 0..383 of qkv)
  __shared__ __bf16 Vt[NH][DKH][40];    // V transposed: [h][d][j], j zero-padded
  __shared__ float  L[NH][32][33];      // qk logits per head
  __shared__ __bf16 Wm[NH][32][40];     // softmax weights, zero-padded
  __shared__ float  DG[NH][20];         // diag terms

  int b = blockIdx.x;
  int tid = threadIdx.x;
  int wave = tid >> 5;
  int lane = tid & 31;
  int h = wave;
  int lh = (lane >> 4) & 1;
  int nl = lane & 15;

  { // stage qkv: q,k rows -> Q (dword copies); v -> Vt transposed (u16 scatter)
    const unsigned* src = (const unsigned*)(qkv + (size_t)b * VN * 576);
    unsigned* qd = (unsigned*)Q;
    unsigned short* vt = (unsigned short*)&Vt[0][0][0];
    for (int idx = tid; idx < NH * DKH * 15; idx += 96) {  // zero pad cols 17..31
      int hd = idx / 15;
      int j  = 17 + (idx - hd * 15);
      vt[hd * 40 + j] = 0;
    }
    for (int idx = tid; idx < VN * 288; idx += 96) {
      int r = idx / 288, c = idx - r * 288;
      unsigned w = src[idx];
      if (c < 192) {
        qd[r * (QLD / 2) + c] = w;
      } else {
        int d = (c - 192) * 2;           // even, both d in same head block
        vt[d * 40 + r]       = (unsigned short)(w & 0xffffu);
        vt[(d + 1) * 40 + r] = (unsigned short)(w >> 16);
      }
    }
  }
  __syncthreads();

  // ---- logits = q @ k^T (padded 32x32x64, 8 WMMAs) ----
  v8f acc00 = {}, acc01 = {}, acc10 = {}, acc11 = {};
#pragma unroll
  for (int ks = 0; ks < 2; ++ks) {
    v16bf a0 = packA_pad(Q + h * DKH + ks * 32, QLD, 0, VN);
    v16bf a1 = packA_pad(Q + h * DKH + ks * 32, QLD, 16, VN);
    v16bf b0 = packBt_pad(Q + DIMC + h * DKH + ks * 32, QLD, 0, VN);
    v16bf b1 = packBt_pad(Q + DIMC + h * DKH + ks * 32, QLD, 16, VN);
    acc00 = wmma_bf16(a0, b0, acc00);
    acc01 = wmma_bf16(a0, b1, acc01);
    acc10 = wmma_bf16(a1, b0, acc10);
    acc11 = wmma_bf16(a1, b1, acc11);
  }
#pragma unroll
  for (int r = 0; r < 8; ++r) {
    int m = lh * 8 + r;
    L[h][m][nl]           = acc00[r] * 0.125f;
    L[h][m][16 + nl]      = acc01[r] * 0.125f;
    L[h][16 + m][nl]      = acc10[r] * 0.125f;
    L[h][16 + m][16 + nl] = acc11[r] * 0.125f;
  }
  __syncthreads();

  // ---- diag[i] = 0.125 * <q_i, key_rel_diag> ----
  if (lane < VN) {
    DG[h][lane] = 0.125f * dot64_bf(Q + lane * QLD + h * DKH, key_rel_diag);
  }
  __syncthreads();

  // ---- rel-logit skew + masks + softmax + drop + renorm (lane = row i) ----
  float wrow[VN];
  if (lane < VN) {
    int i = lane;
    const float* gmr = gmask + (((size_t)b * NH + h) * VN + i) * VN;
    float mx = -3.0e38f;
#pragma unroll 1
    for (int j = 0; j < VN; ++j) {
      int t = i * VN + j;
      int a = t / 18;
      int c = t - a * 18;
      float rl;
      if (c == 0) {
        rl = DG[h][a];
      } else if (a == 16) {
        rl = 0.f;
      } else {
        int idx = a * VN + (c - 1);
        int ap = idx >> 4;
        int jp = idx & 15;
        rl = 0.125f * dot64_bf(Q + ap * QLD + h * DKH,
                               key_rel + (size_t)(ap * 16 + jp) * DKH);
      }
      float lg = L[h][i][j] + mask[i * VN + j] * PA[i * VN + j] + gmr[j] + rl;
      wrow[j] = lg;
      mx = fmaxf(mx, lg);
    }
    float sum = 0.f;
#pragma unroll
    for (int j = 0; j < VN; ++j) { float e = __expf(wrow[j] - mx); wrow[j] = e; sum += e; }
    float inv = 1.f / sum;
    const float* dr = drop + ((size_t)b * NH + h) * VN;
    float s2 = 0.f;
#pragma unroll
    for (int j = 0; j < VN; ++j) { float wd = wrow[j] * inv * dr[j]; wrow[j] = wd; s2 += wd; }
    float inv2 = 1.f / (s2 + 1e-8f);
#pragma unroll
    for (int j = 0; j < VN; ++j) wrow[j] *= inv2;
  }

  { // zero the 32x40 W tile (dword stores), then fill valid 17x17
    unsigned* wp = (unsigned*)&Wm[h][0][0];
    for (int idx = lane; idx < 32 * 20; idx += 32) wp[idx] = 0u;
  }
  __syncthreads();
  if (lane < VN) {
#pragma unroll
    for (int j = 0; j < VN; ++j) Wm[h][lane][j] = (__bf16)wrow[j];
  }
  __syncthreads();

  // ---- out = w @ v (padded 32x64x32, 8 WMMAs); V^T tile feeds packBt ----
  v8f vzero = {};
  v8f oacc[2][4];
#pragma unroll
  for (int mt = 0; mt < 2; ++mt)
#pragma unroll
    for (int nt = 0; nt < 4; ++nt) oacc[mt][nt] = vzero;

  v16bf aw0 = packA(&Wm[h][0][0], 40);
  v16bf aw1 = packA(&Wm[h][16][0], 40);
#pragma unroll
  for (int nt = 0; nt < 4; ++nt) {
    v16bf bv = packBt(&Vt[h][nt * 16][0], 40);
    oacc[0][nt] = wmma_bf16(aw0, bv, oacc[0][nt]);
    oacc[1][nt] = wmma_bf16(aw1, bv, oacc[1][nt]);
  }
#pragma unroll
  for (int mt = 0; mt < 2; ++mt) {
    int mbase = mt * 16 + lh * 8;
#pragma unroll
    for (int r = 0; r < 8; ++r) {
      int mm = mbase + r;
      if (mm < VN) {
#pragma unroll
        for (int nt = 0; nt < 4; ++nt)
          outA[((size_t)b * VN + mm) * DIMC + h * DKH + nt * 16 + nl] =
              (__bf16)oacc[mt][nt][r];
      }
    }
  }
}

// ---------------------------------------------------------------------------
// Fused output GEMMs: Mid = A0 @ attn_w^T + bA (kept in LDS, bf16);
//                     Out = Mid @ nn1_w^T + bN (fp32 to d_out)
// B staged in 96-wide k-groups; A operands preloaded to registers per stage.
// LDS: 25600 + 25600 + 13312 = 64512 B
// ---------------------------------------------------------------------------
__global__ __launch_bounds__(256) void gemm_out_fused_kernel(
    const __bf16* __restrict__ A0, const float* __restrict__ WA,
    const float* __restrict__ bA, const float* __restrict__ WNW,
    const float* __restrict__ bN, float* __restrict__ Out) {
  __shared__ __bf16 As[64][200];
  __shared__ __bf16 Mid[64][200];
  __shared__ __bf16 Bc[64][104];   // [n][k-chunk of 96]
  int tid = threadIdx.x;
  int m0  = blockIdx.x * 64;
  { // 64x192 bf16 tile = 6144 dwords
    const unsigned* src = (const unsigned*)(A0 + (size_t)m0 * DIMC);
#pragma unroll 4
    for (int i = 0; i < 24; ++i) {
      int idx = tid + 256 * i;
      int m = idx / 96;
      int c = idx - m * 96;
      ((unsigned*)&As[m][0])[c] = src[(size_t)m * 96 + c];
    }
  }
  __syncthreads();
  int wave = tid >> 5, lane = tid & 31;
  int wm = wave >> 1, wn = wave & 1;
  int lh = (lane >> 4) & 1, nl = lane & 15;

  v16bf areg[6];
#pragma unroll
  for (int ks = 0; ks < 6; ++ks) areg[ks] = packA(&As[wm * 16][ks * 32], 200);

  // stage 1: Mid = A0 @ WA^T + bA
  for (int nb = 0; nb < 3; ++nb) {
    v8f acc0 = {}, acc1 = {};
#pragma unroll
    for (int g = 0; g < 2; ++g) {
      __syncthreads();
#pragma unroll
      for (int i = 0; i < 6; ++i) {   // 64x96 weights via float4
        int idx = tid + 256 * i;
        int n = idx / 24, kc = idx - n * 24;
        float4 f = ((const float4*)(WA + (size_t)(nb * 64 + n) * 192 + g * 96))[kc];
        *(v4bf*)&Bc[n][kc * 4] = cvt4(f);
      }
      __syncthreads();
#pragma unroll
      for (int k3 = 0; k3 < 3; ++k3) {
        int ks = g * 3 + k3;
        acc0 = wmma_bf16(areg[ks], packBt(&Bc[wn * 32][k3 * 32], 104), acc0);
        acc1 = wmma_bf16(areg[ks], packBt(&Bc[wn * 32 + 16][k3 * 32], 104), acc1);
      }
    }
    int nA = nb * 64 + wn * 32 + nl;
    int mr = wm * 16 + lh * 8;
#pragma unroll
    for (int r = 0; r < 8; ++r) {
      Mid[mr + r][nA]      = (__bf16)(acc0[r] + bA[nA]);
      Mid[mr + r][nA + 16] = (__bf16)(acc1[r] + bA[nA + 16]);
    }
  }
  __syncthreads();

  v16bf mreg[6];
#pragma unroll
  for (int ks = 0; ks < 6; ++ks) mreg[ks] = packA(&Mid[wm * 16][ks * 32], 200);

  // stage 2: Out = Mid @ WN^T + bN
  for (int nb = 0; nb < 3; ++nb) {
    v8f acc0 = {}, acc1 = {};
#pragma unroll
    for (int g = 0; g < 2; ++g) {
      __syncthreads();
#pragma unroll
      for (int i = 0; i < 6; ++i) {
        int idx = tid + 256 * i;
        int n = idx / 24, kc = idx - n * 24;
        float4 f = ((const float4*)(WNW + (size_t)(nb * 64 + n) * 192 + g * 96))[kc];
        *(v4bf*)&Bc[n][kc * 4] = cvt4(f);
      }
      __syncthreads();
#pragma unroll
      for (int k3 = 0; k3 < 3; ++k3) {
        int ks = g * 3 + k3;
        acc0 = wmma_bf16(mreg[ks], packBt(&Bc[wn * 32][k3 * 32], 104), acc0);
        acc1 = wmma_bf16(mreg[ks], packBt(&Bc[wn * 32 + 16][k3 * 32], 104), acc1);
      }
    }
    int gn = nb * 64 + wn * 32 + nl;
    int gm = m0 + wm * 16 + lh * 8;
#pragma unroll
    for (int r = 0; r < 8; ++r) {
      size_t o = (size_t)(gm + r) * DIMC;
      Out[o + gn]      = acc0[r] + bN[gn];
      Out[o + gn + 16] = acc1[r] + bN[gn + 16];
    }
  }
}

// ---------------------------------------------------------------------------
extern "C" void kernel_launch(void* const* d_in, const int* in_sizes, int n_in,
                              void* d_out, int out_size, void* d_ws, size_t ws_size,
                              hipStream_t stream) {
  const float* x            = (const float*)d_in[0];
  const float* mask         = (const float*)d_in[1];
  const float* gmask        = (const float*)d_in[2];
  const float* drop         = (const float*)d_in[3];
  const float* qkv_w        = (const float*)d_in[4];
  const float* qkv_b        = (const float*)d_in[5];
  const float* attn_w       = (const float*)d_in[6];
  const float* attn_b       = (const float*)d_in[7];
  const float* key_rel      = (const float*)d_in[8];
  const float* key_rel_diag = (const float*)d_in[9];
  const float* PA           = (const float*)d_in[10];
  const float* nn1_w        = (const float*)d_in[11];
  const float* nn1_b        = (const float*)d_in[12];

  char* ws = (char*)d_ws;
  size_t off = 0;
  auto wsalloc = [&](size_t bytes) -> void* {
    void* p = ws + off;
    off += (bytes + 255) & ~(size_t)255;
    return p;
  };
  __bf16* qkv_ws  = (__bf16*)wsalloc((size_t)MROWS * 576 * sizeof(__bf16));  // ~160 MB
  __bf16* attn_ws = (__bf16*)wsalloc((size_t)MROWS * DIMC * sizeof(__bf16)); // ~53 MB

  gemm_qkv_kernel<<<MROWS / 64, 256, 0, stream>>>(x, qkv_w, qkv_b, qkv_ws, 576);

  attn_kernel<<<BATCH, 96, 0, stream>>>(qkv_ws, mask, PA, gmask, drop,
                                        key_rel, key_rel_diag, attn_ws);

  gemm_out_fused_kernel<<<MROWS / 64, 256, 0, stream>>>(attn_ws, attn_w, attn_b,
                                                        nn1_w, nn1_b, (float*)d_out);
}
